// GNNActor_65455301591867
// MI455X (gfx1250) — matrix operand order repaired and checked by reading, not compiled, not run on hardware
//
#include <hip/hip_runtime.h>
#include <hip/hip_bf16.h>

// N=100000, E=1600000, IN=10, H=64, T=10 (derived from in_sizes at launch)

typedef __attribute__((ext_vector_type(16))) _Float16 v16h;
typedef __attribute__((ext_vector_type(8)))  _Float16 v8h;
typedef __attribute__((ext_vector_type(8)))  float    v8f;

#define CAT8(a, b) __builtin_shufflevector((a), (b), 0,1,2,3,4,5,6,7,8,9,10,11,12,13,14,15)

__device__ __forceinline__ float softplus_f(float x) {
  return log1pf(expf(-fabsf(x))) + fmaxf(x, 0.f);
}

// order-preserving float atomic max via int max (>=0) / uint min (<0).
// Buffer must be initialized to -inf. Bitwise exact & order independent.
__device__ __forceinline__ void atomicMaxF(float* addr, float v) {
  if (v >= 0.f) atomicMax((int*)addr, __float_as_int(v));
  else          atomicMin((unsigned int*)addr, __float_as_uint(v));
}

// ------------------------- elementwise helpers -------------------------
__global__ void k_fill(float* __restrict__ p, float v, int n) {
  int i = blockIdx.x * 256 + threadIdx.x;
  if (i < n) p[i] = v;
}
__global__ void k_deg(const int* __restrict__ dst, float* __restrict__ deg, int E) {
  int i = blockIdx.x * 256 + threadIdx.x;
  if (i < E) atomicAdd(&deg[dst[i]], 1.f);
}
__global__ void k_rsqrt(float* __restrict__ d, int n) {
  int i = blockIdx.x * 256 + threadIdx.x;
  if (i < n) d[i] = 1.f / sqrtf(d[i]);
}
__global__ void k_relu_inplace(float* __restrict__ p, int n) {
  int i = blockIdx.x * 256 + threadIdx.x;
  if (i < n) p[i] = fmaxf(p[i], 0.f);   // also maps -inf (no edges) -> 0
}

// ------------------------- GCN pieces -------------------------
// hw = h[N,Kin] @ w[Kin,64]
__global__ void k_mm64(const float* __restrict__ h, const float* __restrict__ w,
                       float* __restrict__ out, int N, int Kin) {
  int idx = blockIdx.x * 256 + threadIdx.x;
  if (idx >= N * 64) return;
  int i = idx >> 6, c = idx & 63;
  const float* hr = h + (long)i * Kin;
  float s = 0.f;
  for (int k = 0; k < Kin; ++k) s += hr[k] * w[k * 64 + c];
  out[idx] = s;
}
// hw = concat(h0,h1)[N,128] @ w[128,64]
__global__ void k_mm64_cat(const float* __restrict__ h0, const float* __restrict__ h1,
                           const float* __restrict__ w, float* __restrict__ out, int N) {
  int idx = blockIdx.x * 256 + threadIdx.x;
  if (idx >= N * 64) return;
  int i = idx >> 6, c = idx & 63;
  float s = 0.f;
  const float* r0 = h0 + (long)i * 64;
  const float* r1 = h1 + (long)i * 64;
  for (int k = 0; k < 64; ++k) s += r0[k] * w[k * 64 + c];
  for (int k = 0; k < 64; ++k) s += r1[k] * w[(64 + k) * 64 + c];
  out[idx] = s;
}
// hw3 = h[N,64] @ w[64,10], stored padded [N,16]
__global__ void k_mm10(const float* __restrict__ h, const float* __restrict__ w,
                       float* __restrict__ out, int N) {
  int idx = blockIdx.x * 256 + threadIdx.x;
  if (idx >= N * 16) return;
  int i = idx >> 4, c = idx & 15;
  float s = 0.f;
  if (c < 10) {
    const float* hr = h + (long)i * 64;
    for (int k = 0; k < 64; ++k) s += hr[k] * w[k * 10 + c];
  }
  out[idx] = s;
}
// agg[dst] += hw[src] * dis[src]*dis[dst]; 4 channels per thread (float4)
__global__ void k_scat64(const float* __restrict__ hw, const int* __restrict__ src,
                         const int* __restrict__ dst, const float* __restrict__ dis,
                         float* __restrict__ agg, int E) {
  int idx = blockIdx.x * 256 + threadIdx.x;
  if (idx >= E * 16) return;
  int e = idx >> 4, c4 = (idx & 15) << 2;
  int s = src[e], d = dst[e];
  float nrm = dis[s] * dis[d];
  float4 v = *(const float4*)(hw + (long)s * 64 + c4);
  float* a = agg + (long)d * 64 + c4;
  atomicAdd(a + 0, v.x * nrm);
  atomicAdd(a + 1, v.y * nrm);
  atomicAdd(a + 2, v.z * nrm);
  atomicAdd(a + 3, v.w * nrm);
}
// same, padded 16-wide with 10 valid channels
__global__ void k_scat16(const float* __restrict__ hw, const int* __restrict__ src,
                         const int* __restrict__ dst, const float* __restrict__ dis,
                         float* __restrict__ agg, int E) {
  int idx = blockIdx.x * 256 + threadIdx.x;
  if (idx >= E * 16) return;
  int e = idx >> 4, c = idx & 15;
  if (c >= 10) return;
  int s = src[e], d = dst[e];
  float nrm = dis[s] * dis[d];
  atomicAdd(&agg[(long)d * 16 + c], hw[(long)s * 16 + c] * nrm);
}
// out = relu(agg + hw*dis^2 + b)
__global__ void k_fin64(float* __restrict__ agg, const float* __restrict__ hw,
                        const float* __restrict__ dis, const float* __restrict__ b, int N) {
  int idx = blockIdx.x * 256 + threadIdx.x;
  if (idx >= N * 64) return;
  int i = idx >> 6, c = idx & 63;
  float di = dis[i];
  agg[idx] = fmaxf(agg[idx] + hw[idx] * di * di + b[c], 0.f);
}
__global__ void k_fin16(float* __restrict__ agg, const float* __restrict__ hw,
                        const float* __restrict__ dis, const float* __restrict__ b, int N) {
  int idx = blockIdx.x * 256 + threadIdx.x;
  if (idx >= N * 16) return;
  int i = idx >> 4, c = idx & 15;
  float v = 0.f;
  if (c < 10) {
    float di = dis[i];
    v = fmaxf(agg[idx] + hw[idx] * di * di + b[c], 0.f);
  }
  agg[idx] = v;
}

// ------------------------- EdgeConv (WMMA) -------------------------
// Per block: 8 waves x 16 edges. Layer1: [16x32]f16 @ [32x64]f16 (K pad 30->32).
// Layer2: [16x64] @ [64x64]. Scatter-max f32 results into aggE[N,64].
// Weights stored K-contiguous (transposed) in LDS so every fragment is built
// from 16-byte ds_load_b128 runs instead of scalar u16 loads.
__global__ __launch_bounds__(256) void k_edgeconv(
    const float* __restrict__ x, const float* __restrict__ ea,
    const float* __restrict__ w1, const float* __restrict__ b1,
    const float* __restrict__ w2, const float* __restrict__ b2,
    const int* __restrict__ srcI, const int* __restrict__ dstI,
    float* __restrict__ aggE, int E) {
  __shared__ __align__(16) _Float16 w1sT[64][32];    // [n][k], 4 KB (k 30,31 = 0)
  __shared__ __align__(16) _Float16 w2sT[64][64];    // [n][k], 8 KB
  __shared__ __align__(16) _Float16 aT[8][16][32];   // per-wave A tile, 8 KB
  __shared__ __align__(16) _Float16 h1s[8][16][64];  // per-wave layer1 out, 16 KB

  const int tid = threadIdx.x;
  const int wv = tid >> 5;
  const int lane = tid & 31;
  const int laneHi = lane >> 4;
  const int m = lane & 15;
  const int nLane = lane & 15;

  // stage weights (f32 -> f16), coalesced global reads, transposed LDS writes
  for (int idx = tid; idx < 32 * 64; idx += 256) {
    int k = idx >> 6, n = idx & 63;
    w1sT[n][k] = (k < 30) ? (_Float16)w1[k * 64 + n] : (_Float16)0.f;
  }
  for (int idx = tid; idx < 64 * 64; idx += 256) {
    int k = idx >> 6, n = idx & 63;
    w2sT[n][k] = (_Float16)w2[idx];
  }

  // stage A tile row m: K = [x[dst](10) | x[src](10) | ea(10) | 0 0]
  // each lane owns a contiguous 16-half (32 B) run -> two b128 stores
  const int tileBase = blockIdx.x * 128 + wv * 16;
  int e = tileBase + m; if (e >= E) e = E - 1;       // clamp (results discarded)
  const int s = srcI[e], d = dstI[e];
  _Float16 tmp[16];
  if (laneHi == 0) {                                 // halves 0..15
    #pragma unroll
    for (int k = 0; k < 10; ++k) tmp[k] = (_Float16)x[(long)d * 10 + k];
    #pragma unroll
    for (int k = 0; k < 6; ++k) tmp[10 + k] = (_Float16)x[(long)s * 10 + k];
  } else {                                           // halves 16..31
    #pragma unroll
    for (int k = 0; k < 4; ++k) tmp[k] = (_Float16)x[(long)s * 10 + 6 + k];
    #pragma unroll
    for (int k = 0; k < 10; ++k) tmp[4 + k] = (_Float16)ea[(long)e * 10 + k];
    tmp[14] = (_Float16)0.f; tmp[15] = (_Float16)0.f;
  }
  {
    v8h t0, t1;
    #pragma unroll
    for (int i = 0; i < 8; ++i) { t0[i] = tmp[i]; t1[i] = tmp[8 + i]; }
    v8h* arow = (v8h*)(&aT[wv][m][0]);
    arow[2 * laneHi]     = t0;
    arow[2 * laneHi + 1] = t1;
  }
  __syncthreads();

  // A fragment: runs at half-offsets {laneHi*8, 16+laneHi*8}
  v16h afrag;
  {
    const v8h* arow = (const v8h*)(&aT[wv][m][0]);
    afrag = CAT8(arow[laneHi], arow[2 + laneHi]);
  }

  // ---- layer 1: 4 N-tiles of 16 ----
  for (int n0 = 0; n0 < 4; ++n0) {
    const int N1 = n0 * 16 + nLane;
    const v8h* w1r = (const v8h*)(&w1sT[N1][0]);     // B frag: K runs at {laneHi*8, 16+laneHi*8}
    v16h bfrag = CAT8(w1r[laneHi], w1r[2 + laneHi]);
    v8f c = {};
    c = __builtin_amdgcn_wmma_f32_16x16x32_f16(false, afrag, false, bfrag,
                                               (short)0, c, false, false);
    const float bias = b1[N1];
    #pragma unroll
    for (int r = 0; r < 8; ++r) {
      const int M = r + (laneHi ? 8 : 0);
      float v = c[r] + bias;
      h1s[wv][M][N1] = (_Float16)(v > 0.f ? v : 0.f);
    }
  }
  __syncthreads();

  // ---- layer 2: A = h1 (16x64), two K-chunks ----
  v16h a2[2];
  {
    const v8h* hrow = (const v8h*)(&h1s[wv][m][0]);
    #pragma unroll
    for (int kc = 0; kc < 2; ++kc)
      a2[kc] = CAT8(hrow[kc * 4 + laneHi], hrow[kc * 4 + 2 + laneHi]);
  }

  for (int n0 = 0; n0 < 4; ++n0) {
    const int N1 = n0 * 16 + nLane;
    const v8h* w2r = (const v8h*)(&w2sT[N1][0]);
    v8f c = {};
    #pragma unroll
    for (int kc = 0; kc < 2; ++kc) {
      v16h bfrag = CAT8(w2r[kc * 4 + laneHi], w2r[kc * 4 + 2 + laneHi]);
      c = __builtin_amdgcn_wmma_f32_16x16x32_f16(false, a2[kc], false, bfrag,
                                                 (short)0, c, false, false);
    }
    const float bias = b2[N1];
    #pragma unroll
    for (int r = 0; r < 8; ++r) {
      const int M = r + (laneHi ? 8 : 0);
      const int ee = tileBase + M;
      if (ee < E) {
        const int dd = dstI[ee];
        atomicMaxF(&aggE[(long)dd * 64 + N1], c[r] + bias);
      }
    }
  }
}

// ------------------------- final MLP -------------------------
__global__ void k_l1(const float* __restrict__ x, const float* __restrict__ out3,
                     const float* __restrict__ w, const float* __restrict__ b,
                     float* __restrict__ t1, int N) {
  int idx = blockIdx.x * 256 + threadIdx.x;
  if (idx >= N * 16) return;
  int i = idx >> 4, c = idx & 15;
  float v = 0.f;
  if (c < 10) {
    float s = b[c];
    for (int k = 0; k < 10; ++k) s += x[(long)i * 10 + k] * w[k * 10 + c];
    for (int k = 0; k < 10; ++k) s += out3[(long)i * 16 + k] * w[(10 + k) * 10 + c];
    v = softplus_f(s);
  }
  t1[idx] = v;
}
__global__ void k_l2(const float* __restrict__ t1, const float* __restrict__ w,
                     const float* __restrict__ b, float* __restrict__ t2, int N) {
  int idx = blockIdx.x * 256 + threadIdx.x;
  if (idx >= N * 128) return;
  int i = idx >> 7, c = idx & 127;
  float s = b[c];
  for (int k = 0; k < 10; ++k) s += t1[(long)i * 16 + k] * w[k * 128 + c];
  t2[idx] = softplus_f(s);
}
__global__ void k_l3(const float* __restrict__ t2, const float* __restrict__ w,
                     const float* __restrict__ b, float* __restrict__ t3, int N) {
  int idx = blockIdx.x * 256 + threadIdx.x;
  if (idx >= N * 32) return;
  int i = idx >> 5, c = idx & 31;
  float s = b[c];
  const float* r = t2 + (long)i * 128;
  for (int k = 0; k < 128; ++k) s += r[k] * w[k * 32 + c];
  t3[idx] = softplus_f(s);
}
__global__ void k_l4(const float* __restrict__ t3, const float* __restrict__ w,
                     const float* __restrict__ b, float* __restrict__ out, int N) {
  int i = blockIdx.x * 256 + threadIdx.x;
  if (i >= N) return;
  float s0 = b[0], s1 = b[1];
  const float* r = t3 + (long)i * 32;
  for (int k = 0; k < 32; ++k) { float t = r[k]; s0 += t * w[k * 2]; s1 += t * w[k * 2 + 1]; }
  out[i] = s0;
  out[N + i] = s1;
}

extern "C" void kernel_launch(void* const* d_in, const int* in_sizes, int n_in,
                              void* d_out, int out_size, void* d_ws, size_t ws_size,
                              hipStream_t stream) {
  const float* x      = (const float*)d_in[0];
  const float* eattr  = (const float*)d_in[1];
  const float* gcn1_w = (const float*)d_in[2];  const float* gcn1_b = (const float*)d_in[3];
  const float* e_w1   = (const float*)d_in[4];  const float* e_b1   = (const float*)d_in[5];
  const float* e_w2   = (const float*)d_in[6];  const float* e_b2   = (const float*)d_in[7];
  const float* gcn2_w = (const float*)d_in[8];  const float* gcn2_b = (const float*)d_in[9];
  const float* gcn3_w = (const float*)d_in[10]; const float* gcn3_b = (const float*)d_in[11];
  const float* l1_w   = (const float*)d_in[12]; const float* l1_b   = (const float*)d_in[13];
  const float* l2_w   = (const float*)d_in[14]; const float* l2_b   = (const float*)d_in[15];
  const float* l3_w   = (const float*)d_in[16]; const float* l3_b   = (const float*)d_in[17];
  const float* l4_w   = (const float*)d_in[18]; const float* l4_b   = (const float*)d_in[19];
  const int*   ei     = (const int*)d_in[20];

  const int N = in_sizes[0] / 10;
  const int E = in_sizes[20] / 2;
  const int* srcI = ei;
  const int* dstI = ei + E;

  // workspace carve: dis[N] | bufA[N*128] | bufB[N*64] | bufC[N*64]
  float* dis  = (float*)d_ws;
  float* bufA = dis + N;
  float* bufB = bufA + (size_t)N * 128;
  float* bufC = bufB + (size_t)N * 64;

  const dim3 B(256);
  auto g = [](long n) { return dim3((unsigned)((n + 255) / 256)); };

  // degrees -> dis = deg^-0.5 (self loop => init 1)
  k_fill  <<<g(N), B, 0, stream>>>(dis, 1.f, N);
  k_deg   <<<g(E), B, 0, stream>>>(dstI, dis, E);
  k_rsqrt <<<g(N), B, 0, stream>>>(dis, N);

  // --- GCN1: out1 = relu(gcn(x, gcn1_w)) -> bufB ---
  k_mm64  <<<g((long)N * 64), B, 0, stream>>>(x, gcn1_w, bufA, N, 10);
  k_fill  <<<g((long)N * 64), B, 0, stream>>>(bufB, 0.f, N * 64);
  k_scat64<<<g((long)E * 16), B, 0, stream>>>(bufA, srcI, dstI, dis, bufB, E);
  k_fin64 <<<g((long)N * 64), B, 0, stream>>>(bufB, bufA, dis, gcn1_b, N);

  // --- EdgeConv (WMMA f16) -> oute in bufC ---
  k_fill        <<<g((long)N * 64), B, 0, stream>>>(bufC, -INFINITY, N * 64);
  k_edgeconv    <<<dim3((unsigned)((E + 127) / 128)), B, 0, stream>>>(
      x, eattr, e_w1, e_b1, e_w2, e_b2, srcI, dstI, bufC, E);
  k_relu_inplace<<<g((long)N * 64), B, 0, stream>>>(bufC, N * 64);

  // --- GCN2: out2 = relu(gcn(cat(out1,oute), gcn2_w)) -> bufB ---
  k_mm64_cat<<<g((long)N * 64), B, 0, stream>>>(bufB, bufC, gcn2_w, bufA, N);
  k_fill    <<<g((long)N * 64), B, 0, stream>>>(bufB, 0.f, N * 64);
  k_scat64  <<<g((long)E * 16), B, 0, stream>>>(bufA, srcI, dstI, dis, bufB, E);
  k_fin64   <<<g((long)N * 64), B, 0, stream>>>(bufB, bufA, dis, gcn2_b, N);

  // --- GCN3: out3 = relu(gcn(out2, gcn3_w)) padded [N,16] -> bufA ---
  k_mm10  <<<g((long)N * 16), B, 0, stream>>>(bufB, gcn3_w, bufC, N);
  k_fill  <<<g((long)N * 16), B, 0, stream>>>(bufA, 0.f, N * 16);
  k_scat16<<<g((long)E * 16), B, 0, stream>>>(bufC, srcI, dstI, dis, bufA, E);
  k_fin16 <<<g((long)N * 16), B, 0, stream>>>(bufA, bufC, dis, gcn3_b, N);

  // --- MLP head ---
  k_l1<<<g((long)N * 16),  B, 0, stream>>>(x, bufA, l1_w, l1_b, bufC, N);
  k_l2<<<g((long)N * 128), B, 0, stream>>>(bufC, l2_w, l2_b, bufA, N);
  k_l3<<<g((long)N * 32),  B, 0, stream>>>(bufA, l3_w, l3_b, bufB, N);
  k_l4<<<g(N),             B, 0, stream>>>(bufB, l4_w, l4_b, (float*)d_out, N);
}
